// GlobalAttentionPooling_22651657519254
// MI455X (gfx1250) — compile-verified, matrix operand order checked by reading, not run on hardware
//
#include <hip/hip_runtime.h>
#include <hip/hip_bf16.h>

typedef __attribute__((ext_vector_type(16))) _Float16 v16h;
typedef __attribute__((ext_vector_type(8)))  _Float16 v8h;
typedef __attribute__((ext_vector_type(8)))  float    v8f;

#define NN 262144   // nodes
#define DD 256      // feature dim
#define HH 128      // hidden dim (D/2)
#define GG 2048     // graphs

__device__ __forceinline__ int lower_bound_i32(const int* __restrict__ arr, int n, int key) {
    int lo = 0, hi = n;
    while (lo < hi) {
        int mid = (lo + hi) >> 1;
        if (arr[mid] < key) lo = mid + 1; else hi = mid;
    }
    return lo;
}

// ---------------------------------------------------------------------------
// Kernel 1: scores = relu(x @ W1 + b1) @ W2 + b2, using v_wmma_f32_16x16x32_f16
// Block = 256 threads = 8 wave32s; each wave owns a 16-row tile of x.
// LDS: [0,64KB)  W1 in WMMA B-fragment order (f16)
//      [64,128KB) 128 rows of x as f16, row-major
// ---------------------------------------------------------------------------
__global__ __launch_bounds__(256)
void attn_scores_kernel(const float* __restrict__ x,
                        const float* __restrict__ W1,
                        const float* __restrict__ b1,
                        const float* __restrict__ W2,
                        const float* __restrict__ b2,
                        float* __restrict__ scores) {
    extern __shared__ char smem[];
    _Float16* w1f   = (_Float16*)smem;              // 32768 f16 = 64 KB
    _Float16* xtile = (_Float16*)(smem + 65536);    // 128*256 f16 = 64 KB

    const int tid = threadIdx.x;

    // Stage W1 (f32 row-major [k][col]) into B-fragment layout:
    // frag[(ntile*8 + ks)*32 + h*16 + n][e], per ISA 16-bit 32x16 layout.
    for (int idx = tid; idx < DD * HH; idx += 256) {
        const int k   = idx >> 7;            // / HH
        const int col = idx & (HH - 1);
        const int nt  = col >> 4, n  = col & 15;
        const int ks  = k >> 5,  kk = k & 31;
        const int h   = (kk >> 3) & 1;                    // which K-half group
        const int e   = (kk & 7) | ((kk >> 4) << 3);      // element within lane
        w1f[((((nt << 3) + ks) << 5) + (h << 4) + n) * 16 + e] = (_Float16)W1[idx];
    }
    // Stage 128 rows of x, f32 -> f16, fully coalesced.
    const long rowbase = (long)blockIdx.x * 128;
    const float* __restrict__ xsrc = x + rowbase * DD;
    for (int idx = tid; idx < 128 * DD; idx += 256) {
        xtile[idx] = (_Float16)xsrc[idx];
    }
    __syncthreads();

    const int wave = tid >> 5;      // wave32
    const int lane = tid & 31;
    const int h = lane >> 4;        // half-wave
    const int m = lane & 15;
    const int arow = wave * 16 + m; // this lane's A-matrix row within the block

    float sacc[8];
    #pragma unroll
    for (int r = 0; r < 8; ++r) sacc[r] = 0.0f;

    #pragma unroll
    for (int nt = 0; nt < 8; ++nt) {            // 8 x 16 hidden columns
        v8f acc = {};
        #pragma unroll
        for (int ks = 0; ks < 8; ++ks) {        // K = 256 in steps of 32
            const int kb = ks << 5;
            // A fragment: K = kb + h*8 .. +7  and  kb + 16 + h*8 .. +7
            v8h a0 = *(const v8h*)&xtile[arow * DD + kb + (h << 3)];
            v8h a1 = *(const v8h*)&xtile[arow * DD + kb + 16 + (h << 3)];
            v16h a = __builtin_shufflevector(a0, a1,
                        0,1,2,3,4,5,6,7,8,9,10,11,12,13,14,15);
            // B fragment: one contiguous 32-byte LDS load per lane
            v16h b = *(const v16h*)&w1f[((((nt << 3) + ks) << 5) + lane) * 16];
            acc = __builtin_amdgcn_wmma_f32_16x16x32_f16(
                      false, a, false, b, (short)0, acc, false, false);
        }
        // Fused bias + ReLU + dot with W2 column segment.
        // C layout: lane (h,m) holds (M = r + 8h, N = m) in acc[r].
        const int col = (nt << 4) + m;
        const float b1v = b1[col];
        const float w2v = W2[col];
        #pragma unroll
        for (int r = 0; r < 8; ++r) {
            float hv = acc[r] + b1v;
            hv = hv > 0.0f ? hv : 0.0f;
            sacc[r] = fmaf(hv, w2v, sacc[r]);
        }
    }

    // Reduce partial scores across the 16 lanes of each half-wave.
    const float bias2 = b2[0];
    #pragma unroll
    for (int r = 0; r < 8; ++r) {
        float v = sacc[r];
        v += __shfl_xor(v, 1, 32);
        v += __shfl_xor(v, 2, 32);
        v += __shfl_xor(v, 4, 32);
        v += __shfl_xor(v, 8, 32);
        sacc[r] = v + bias2;
    }
    if (m == 0) {   // lane 0 writes rows 0..7, lane 16 writes rows 8..15
        const long row0 = rowbase + wave * 16 + (h << 3);
        #pragma unroll
        for (int r = 0; r < 8; ++r) scores[row0 + r] = sacc[r];
    }
}

// ---------------------------------------------------------------------------
// Kernel 2: per-graph max and exp-sum (stable segment softmax stats).
// ---------------------------------------------------------------------------
__global__ __launch_bounds__(256)
void seg_stats_kernel(const float* __restrict__ scores,
                      const int* __restrict__ batch,
                      float* __restrict__ segmax,
                      float* __restrict__ segsum) {
    const int g = blockIdx.x;
    const int tid = threadIdx.x;
    __shared__ float red[8];
    __shared__ float bcast;
    __shared__ int range[2];
    if (tid == 0) {
        range[0] = lower_bound_i32(batch, NN, g);
        range[1] = lower_bound_i32(batch, NN, g + 1);
    }
    __syncthreads();
    const int s = range[0], e = range[1];

    float m = -3.402823466e38f;
    for (int i = s + tid; i < e; i += 256) m = fmaxf(m, scores[i]);
    #pragma unroll
    for (int mask = 16; mask >= 1; mask >>= 1) m = fmaxf(m, __shfl_xor(m, mask, 32));
    if ((tid & 31) == 0) red[tid >> 5] = m;
    __syncthreads();
    if (tid == 0) {
        float mm = red[0];
        #pragma unroll
        for (int w = 1; w < 8; ++w) mm = fmaxf(mm, red[w]);
        bcast = mm;
    }
    __syncthreads();
    const float gmax = bcast;

    float sum = 0.0f;
    for (int i = s + tid; i < e; i += 256) sum += __expf(scores[i] - gmax);
    #pragma unroll
    for (int mask = 16; mask >= 1; mask >>= 1) sum += __shfl_xor(sum, mask, 32);
    if ((tid & 31) == 0) red[tid >> 5] = sum;
    __syncthreads();
    if (tid == 0) {
        float ss = 0.0f;
        #pragma unroll
        for (int w = 0; w < 8; ++w) ss += red[w];
        segmax[g] = gmax;
        segsum[g] = ss;
    }
}

// ---------------------------------------------------------------------------
// Kernel 3: softmax weights per node.
// ---------------------------------------------------------------------------
__global__ __launch_bounds__(256)
void weights_kernel(const float* __restrict__ scores,
                    const int* __restrict__ batch,
                    const float* __restrict__ segmax,
                    const float* __restrict__ segsum,
                    float* __restrict__ weights) {
    const int i = blockIdx.x * 256 + threadIdx.x;
    if (i < NN) {
        const int g = batch[i];
        weights[i] = __expf(scores[i] - segmax[g]) / segsum[g];
    }
}

// ---------------------------------------------------------------------------
// Kernel 4: weighted pooling. One block per graph, thread d owns column d.
// batch is sorted -> each graph is a contiguous node range, no atomics.
// ---------------------------------------------------------------------------
__global__ __launch_bounds__(256)
void pool_kernel(const float* __restrict__ x,
                 const int* __restrict__ batch,
                 const float* __restrict__ weights,
                 float* __restrict__ out) {
    const int g = blockIdx.x;
    const int d = threadIdx.x;
    __shared__ int range[2];
    if (d == 0) {
        range[0] = lower_bound_i32(batch, NN, g);
        range[1] = lower_bound_i32(batch, NN, g + 1);
    }
    __syncthreads();
    const int s = range[0], e = range[1];
    float acc = 0.0f;
    for (int i = s; i < e; ++i) {
        if (i + 2 < e) __builtin_prefetch(&x[(long)(i + 2) * DD + d], 0, 0);
        acc = fmaf(x[(long)i * DD + d], weights[i], acc);
    }
    out[(long)g * DD + d] = acc;
}

// ---------------------------------------------------------------------------
extern "C" void kernel_launch(void* const* d_in, const int* in_sizes, int n_in,
                              void* d_out, int out_size, void* d_ws, size_t ws_size,
                              hipStream_t stream) {
    const float* x     = (const float*)d_in[0];
    const int*   batch = (const int*)d_in[1];
    // d_in[2] = num_graphs (compile-time GG here)
    const float* W1 = (const float*)d_in[3];
    const float* b1 = (const float*)d_in[4];
    const float* W2 = (const float*)d_in[5];
    const float* b2 = (const float*)d_in[6];
    float* out = (float*)d_out;

    float* scores  = (float*)d_ws;          // NN floats
    float* weights = scores + NN;           // NN floats
    float* segmax  = weights + NN;          // GG floats
    float* segsum  = segmax + GG;           // GG floats

    attn_scores_kernel<<<NN / 128, 256, 128 * 1024, stream>>>(x, W1, b1, W2, b2, scores);
    seg_stats_kernel<<<GG, 256, 0, stream>>>(scores, batch, segmax, segsum);
    weights_kernel<<<(NN + 255) / 256, 256, 0, stream>>>(scores, batch, segmax, segsum, weights);
    pool_kernel<<<GG, 256, 0, stream>>>(x, batch, weights, out);
}
